// RankEnhancingTokenFuser_37005438222683
// MI455X (gfx1250) — matrix operand order repaired and checked by reading, not compiled, not run on hardware
//
#include <hip/hip_runtime.h>
#include <hip/hip_bf16.h>

// ---------------------------------------------------------------------------
// RankEnhancingTokenFuser for MI455X (gfx1250, wave32, WMMA + TDM)
//  K0 alpha     : sigmoid of per-channel logits
//  K1 pool_mu   : masked pooling over L, column means of first 256 rows
//  K2 rdot      : r_i = x_i . mu  (i=256 -> mu.mu)   [centering correction]
//  K3 gram      : A = X Xt via v_wmma_f32_16x16x32_f16 (f16 hi/lo split);
//                 tiles staged LDS-side by tensor_load_to_lds (TDM)
//  K4 subspace  : 6x (Z = A*Q ; MGS) -> orthonormal top-64 basis Q
//  K5 importance: I[d] = ||Qt (x_d - mu)||^2 as WMMA GEMM + squared reduce
//  K6 select    : radix-select 410 smallest I -> mask
//  K7 fuse      : out = x + m * alpha * (pool_other - x)   (non-temporal)
// ---------------------------------------------------------------------------

#define Bn   16
#define LT   512
#define LI   576
#define Dn   4096
#define Sn   256     // SUBSAMPLE rows
#define Qr   64      // TOP_Q rank
#define KLOW 410     // round(0.1 * 4096)

typedef _Float16 v8h  __attribute__((ext_vector_type(8)));
typedef _Float16 v16h __attribute__((ext_vector_type(16)));
typedef float    v8f  __attribute__((ext_vector_type(8)));
typedef float    v4f  __attribute__((ext_vector_type(4)));

#define WMMA_F16(A_, B_, C_) \
  __builtin_amdgcn_wmma_f32_16x16x32_f16(false, (A_), false, (B_), (short)0, (C_), false, false)

// --------------------- TDM: 2D f32 tile -> LDS (D# per ISA §8) -------------
__device__ __forceinline__ void retf_tdm_load_2d(unsigned lds_addr, const void* gaddr,
                                                 int tile_x, int tile_y,
                                                 int tensor_x, int tensor_y,
                                                 long stride_x) {
  typedef unsigned int u32x4 __attribute__((ext_vector_type(4)));
  typedef int i32x8 __attribute__((ext_vector_type(8)));
  typedef int i32x4 __attribute__((ext_vector_type(4)));
  unsigned long long ga = (unsigned long long)gaddr;
  u32x4 g0;
  g0[0] = 1u;                                              // count=1, user mode
  g0[1] = lds_addr;                                        // LDS byte address
  g0[2] = (unsigned)(ga & 0xffffffffu);                    // global_addr[31:0]
  g0[3] = (unsigned)((ga >> 32) & 0x01ffffffu) | (2u << 30); // [56:32] | type=2
  i32x8 g1;
  g1[0] = (int)(2u << 16);                                 // data_size = 4B
  g1[1] = (int)(((unsigned)tensor_x & 0xffffu) << 16);     // tensor_dim0[15:0]
  g1[2] = (int)((((unsigned)tensor_x >> 16) & 0xffffu) |
                (((unsigned)tensor_y & 0xffffu) << 16));   // dim0 hi | dim1 lo
  g1[3] = (int)((((unsigned)tensor_y >> 16) & 0xffffu) |
                (((unsigned)tile_x & 0xffffu) << 16));     // dim1 hi | tile_dim0
  g1[4] = (int)((unsigned)tile_y & 0xffffu);               // tile_dim1, tile_dim2=0
  g1[5] = (int)((unsigned)((unsigned long long)stride_x & 0xffffffffu)); // stride0 lo
  g1[6] = (int)((unsigned)(((unsigned long long)stride_x >> 32) & 0xffffu)); // stride0 hi
  g1[7] = 0;
  i32x4 gz = {0, 0, 0, 0};
#if __clang_major__ >= 23
  i32x8 gz8 = {0, 0, 0, 0, 0, 0, 0, 0};
  __builtin_amdgcn_tensor_load_to_lds(g0, g1, gz, gz, gz8, 0);
#else
  __builtin_amdgcn_tensor_load_to_lds(g0, g1, gz, gz, 0);
#endif
}

// ------------------------------- K0: alpha ---------------------------------
__global__ void retf_alpha_kernel(const float* __restrict__ txt,
                                  const float* __restrict__ img,
                                  float* __restrict__ alpha) {
  int d = blockIdx.x * 256 + threadIdx.x;
  int mod = blockIdx.y;                       // 0 = text, 1 = image
  const float* src = mod ? img : txt;
  alpha[(size_t)mod * Dn + d] = 1.0f / (1.0f + __expf(-src[d]));
}

// ---------------------------- K1: pools + means ----------------------------
__global__ __launch_bounds__(256)
void retf_pool_mu_kernel(const float* __restrict__ X,
                         const unsigned char* __restrict__ mask,
                         int L, float* __restrict__ pool, float* __restrict__ mu) {
  int d = blockIdx.x * 256 + threadIdx.x;
  int b = blockIdx.y;
  const float* Xb = X + (size_t)b * L * Dn + d;
  float ps = 0.f, ms = 0.f, cnt = 0.f;
  for (int l = 0; l < L; ++l) {
    if (l + 16 < L) __builtin_prefetch(Xb + (size_t)(l + 16) * Dn, 0, 0);
    float v  = Xb[(size_t)l * Dn];
    float mf = mask ? (mask[b * L + l] ? 1.f : 0.f) : 1.f;
    ps += v * mf;
    cnt += mf;
    if (l < Sn) ms += v;
  }
  pool[(size_t)b * Dn + d] = ps / fmaxf(cnt, 1.f);
  mu  [(size_t)b * Dn + d] = ms * (1.f / (float)Sn);
}

// ------------------------------- K2: r dots --------------------------------
__global__ __launch_bounds__(256)
void retf_rdot_kernel(const float* __restrict__ Tx, const float* __restrict__ Ix,
                      const float* __restrict__ mu, float* __restrict__ rr) {
  int i = blockIdx.x;            // 0..256  (256 -> mu.mu)
  int b = blockIdx.y;
  int mod = blockIdx.z;
  const float* X = mod ? Ix : Tx;
  int L = mod ? LI : LT;
  const float* mub = mu + (size_t)(mod * Bn + b) * Dn;
  const float* row = (i < Sn) ? (X + ((size_t)b * L + i) * Dn) : mub;
  float s = 0.f;
  for (int d = threadIdx.x; d < Dn; d += 256) s += row[d] * mub[d];
  __shared__ float sc[9];
  int lane = threadIdx.x & 31, w = threadIdx.x >> 5;
#pragma unroll
  for (int o = 16; o > 0; o >>= 1) s += __shfl_xor(s, o, 32);
  if (lane == 0) sc[w] = s;
  __syncthreads();
  if (threadIdx.x == 0) {
    float tot = 0.f;
    for (int q = 0; q < 8; ++q) tot += sc[q];
    rr[(size_t)(mod * Bn + b) * 257 + i] = tot;
  }
}

// ----------------------- fragment loaders (ISA layouts) --------------------
// A: lane L holds row (L&15); K = 8*(L>>4)+[0..7] (elems 0-7), +16 (elems 8-15)
__device__ __forceinline__ v16h retf_frag_a(const _Float16 (*s)[40], int row, int kh) {
  const _Float16* p = &s[row][kh << 3];
  v8h x = *(const v8h*)p;
  v8h y = *(const v8h*)(p + 16);
  return __builtin_shufflevector(x, y, 0,1,2,3,4,5,6,7,8,9,10,11,12,13,14,15);
}
// B: lane L holds column (L&15); K = 8*(L>>4)+[0..7] (elems 0-7), +16 (elems 8-15)
__device__ __forceinline__ v16h retf_frag_b(const _Float16 (*s)[40], int col, int kh) {
  const _Float16* p = &s[col][kh << 3];
  v8h x = *(const v8h*)p;
  v8h y = *(const v8h*)(p + 16);
  return __builtin_shufflevector(x, y, 0,1,2,3,4,5,6,7,8,9,10,11,12,13,14,15);
}

// --------------------- K3: Gram via WMMA, TDM staging ----------------------
__global__ __launch_bounds__(256)
void retf_gram_kernel(const float* __restrict__ Tx, const float* __restrict__ Ix,
                      const float* __restrict__ rr, float* __restrict__ G) {
  int reg = blockIdx.x;          // 0..15 : 4x4 regions of 64x64
  int b   = blockIdx.y;
  int mod = blockIdx.z;
  const float* X = mod ? Ix : Tx;
  int L = mod ? LI : LT;
  int i0 = (reg >> 2) * 64, j0 = (reg & 3) * 64;

  __shared__ float sRawA[64][32];
  __shared__ float sRawB[64][32];
  __shared__ _Float16 sAhi[64][40], sAlo[64][40], sBhi[64][40], sBlo[64][40];

  int tid = threadIdx.x, lane = tid & 31, w = tid >> 5;
  int t0 = 2 * w;
  int ti = t0 >> 2, tj0 = t0 & 3, tj1 = tj0 + 1;
  int m = lane & 15, kh = lane >> 4;

  v8f acc0 = {0.f,0.f,0.f,0.f,0.f,0.f,0.f,0.f};
  v8f acc1 = {0.f,0.f,0.f,0.f,0.f,0.f,0.f,0.f};
  const float* Xb = X + (size_t)b * L * Dn;
  const float* baseA = Xb + (size_t)i0 * Dn;
  const float* baseB = Xb + (size_t)j0 * Dn;
  unsigned ldsA = (unsigned)(size_t)(void*)&sRawA[0][0];
  unsigned ldsB = (unsigned)(size_t)(void*)&sRawB[0][0];

  int cr = tid >> 2;             // conversion row   0..63
  int cc = (tid & 3) << 3;       // conversion col   0,8,16,24

  for (int kc = 0; kc < Dn; kc += 32) {
    __syncthreads();
    if (tid == 0) {
      retf_tdm_load_2d(ldsA, baseA + kc, 32, 64, Dn, Sn, (long)Dn);
      retf_tdm_load_2d(ldsB, baseB + kc, 32, 64, Dn, Sn, (long)Dn);
      __builtin_amdgcn_s_wait_tensorcnt((short)0);
    }
    __syncthreads();
#pragma unroll
    for (int e = 0; e < 8; ++e) {
      float f = sRawA[cr][cc + e];
      _Float16 h = (_Float16)f;
      sAhi[cr][cc + e] = h;
      sAlo[cr][cc + e] = (_Float16)(f - (float)h);
      f = sRawB[cr][cc + e];
      h = (_Float16)f;
      sBhi[cr][cc + e] = h;
      sBlo[cr][cc + e] = (_Float16)(f - (float)h);
    }
    __syncthreads();

    v16h ahi = retf_frag_a(sAhi, ti * 16 + m, kh);
    v16h alo = retf_frag_a(sAlo, ti * 16 + m, kh);
    v16h b0h = retf_frag_b(sBhi, tj0 * 16 + m, kh);
    v16h b0l = retf_frag_b(sBlo, tj0 * 16 + m, kh);
    v16h b1h = retf_frag_b(sBhi, tj1 * 16 + m, kh);
    v16h b1l = retf_frag_b(sBlo, tj1 * 16 + m, kh);

    // f32-via-f16 split: hi*hi + hi*lo + lo*hi (lo*lo below rounding)
    acc0 = WMMA_F16(ahi, b0h, acc0);
    acc0 = WMMA_F16(ahi, b0l, acc0);
    acc0 = WMMA_F16(alo, b0h, acc0);
    acc1 = WMMA_F16(ahi, b1h, acc1);
    acc1 = WMMA_F16(ahi, b1l, acc1);
    acc1 = WMMA_F16(alo, b1h, acc1);
  }

  const float* rb = rr + (size_t)(mod * Bn + b) * 257;
  float mumu = rb[256];
  float* Gb = G + (size_t)(mod * Bn + b) * (Sn * Sn);
  int gj0 = j0 + tj0 * 16 + m;
  int gj1 = j0 + tj1 * 16 + m;
#pragma unroll
  for (int r8 = 0; r8 < 8; ++r8) {
    int gi = i0 + ti * 16 + r8 + 8 * kh;      // C/D layout: M = r + 8*(lane>>4)
    float ri = rb[gi];
    Gb[(size_t)gi * Sn + gj0] = acc0[r8] - ri - rb[gj0] + mumu;
    Gb[(size_t)gi * Sn + gj1] = acc1[r8] - ri - rb[gj1] + mumu;
  }
}

// ------------------- K4: block power iteration + MGS -----------------------
__global__ __launch_bounds__(256)
void retf_subspace_kernel(const float* __restrict__ G, float* __restrict__ Qout) {
  int b = blockIdx.x, mod = blockIdx.y;
  const float* A = G + (size_t)(mod * Bn + b) * (Sn * Sn);
  float* Qb = Qout + (size_t)(mod * Bn + b) * (Sn * Qr);

  __shared__ float sQ[Sn][Qr + 4];   // stride 68 floats: 16B aligned rows
  __shared__ float sPart[Qr][8];
  __shared__ float sDot[Qr];
  __shared__ float sNrm[9];

  int t = threadIdx.x, lane = t & 31, w = t >> 5;

  for (int e = 0; e < Qr; ++e) sQ[t][e] = (t == e) ? 1.f : 0.f;
  __syncthreads();

  for (int it = 0; it < 6; ++it) {
    float zn[Qr];
#pragma unroll
    for (int e = 0; e < Qr; ++e) zn[e] = 0.f;
    for (int j = 0; j < Sn; ++j) {
      float a = A[(size_t)j * Sn + t];        // A symmetric -> coalesced
#pragma unroll
      for (int e4 = 0; e4 < Qr / 4; ++e4) {
        v4f q = *(const v4f*)&sQ[j][e4 << 2];
        zn[e4 * 4 + 0] += a * q.x;
        zn[e4 * 4 + 1] += a * q.y;
        zn[e4 * 4 + 2] += a * q.z;
        zn[e4 * 4 + 3] += a * q.w;
      }
    }
    __syncthreads();
#pragma unroll
    for (int e = 0; e < Qr; ++e) sQ[t][e] = zn[e];
    __syncthreads();

    for (int c = 0; c < Qr; ++c) {
      float v = sQ[t][c];
      float p = v * v;
#pragma unroll
      for (int o = 16; o > 0; o >>= 1) p += __shfl_xor(p, o, 32);
      if (lane == 0) sNrm[w] = p;
      __syncthreads();
      if (t == 0) {
        float s = 0.f;
        for (int q = 0; q < 8; ++q) s += sNrm[q];
        sNrm[8] = s;
      }
      __syncthreads();
      float inv = rsqrtf(fmaxf(sNrm[8], 1e-30f));
      v *= inv;
      sQ[t][c] = v;
      for (int e = c + 1; e < Qr; ++e) {
        float pe = v * sQ[t][e];
#pragma unroll
        for (int o = 16; o > 0; o >>= 1) pe += __shfl_xor(pe, o, 32);
        if (lane == 0) sPart[e][w] = pe;
      }
      __syncthreads();
      if (t > c && t < Qr) {
        float s = 0.f;
        for (int q = 0; q < 8; ++q) s += sPart[t][q];
        sDot[t] = s;
      }
      __syncthreads();
      for (int e = c + 1; e < Qr; ++e) sQ[t][e] -= sDot[e] * v;
      __syncthreads();
    }
  }
  for (int e = 0; e < Qr; ++e) Qb[(size_t)t * Qr + e] = sQ[t][e];
}

// ---------------- K5: channel importance as WMMA GEMM ----------------------
// P = Qt (64 x 256) * Xc (256 x 64-chunk), then I[d] = sum_q P[q,d]^2
__global__ __launch_bounds__(256)
void retf_importance_kernel(const float* __restrict__ Tx, const float* __restrict__ Ix,
                            const float* __restrict__ mu, const float* __restrict__ Qg,
                            float* __restrict__ Iout) {
  int d0 = blockIdx.x * 64;
  int b = blockIdx.y, mod = blockIdx.z;
  const float* X = mod ? Ix : Tx;
  int L = mod ? LI : LT;
  const float* mub = mu + (size_t)(mod * Bn + b) * Dn + d0;
  const float* Qb  = Qg + (size_t)(mod * Bn + b) * (Sn * Qr);
  const float* Xb  = X + (size_t)b * L * Dn + d0;

  __shared__ _Float16 sQhi[64][40], sQlo[64][40], sXhi[64][40], sXlo[64][40];
  __shared__ float sAcc[64][8];

  int tid = threadIdx.x, lane = tid & 31, w = tid >> 5;
  int t0 = 2 * w;
  int ti = t0 >> 2, tj0 = t0 & 3, tj1 = tj0 + 1;
  int m = lane & 15, kh = lane >> 4;

  v8f acc0 = {0.f,0.f,0.f,0.f,0.f,0.f,0.f,0.f};
  v8f acc1 = {0.f,0.f,0.f,0.f,0.f,0.f,0.f,0.f};

  int kk = tid >> 3;             // 0..31 : row within K chunk
  int e8 = (tid & 7) << 3;       // 0..56 : q / d offset

  for (int ic = 0; ic < Sn; ic += 32) {
    __syncthreads();
    // stage Qt (transpose [i][q] -> [q][i]) and centered X ([i][d] -> [d][i])
    const float* qp = Qb + (size_t)(ic + kk) * Qr + e8;
    v4f qa  = *(const v4f*)qp;
    v4f qb4 = *(const v4f*)(qp + 4);
    const float* xp = Xb + (size_t)(ic + kk) * Dn + e8;
    v4f xa  = *(const v4f*)xp;
    v4f xb4 = *(const v4f*)(xp + 4);
#pragma unroll
    for (int e = 0; e < 4; ++e) {
      float f = qa[e];
      _Float16 h = (_Float16)f;
      sQhi[e8 + e][kk] = h; sQlo[e8 + e][kk] = (_Float16)(f - (float)h);
      f = qb4[e];
      h = (_Float16)f;
      sQhi[e8 + 4 + e][kk] = h; sQlo[e8 + 4 + e][kk] = (_Float16)(f - (float)h);
      f = xa[e] - mub[e8 + e];
      h = (_Float16)f;
      sXhi[e8 + e][kk] = h; sXlo[e8 + e][kk] = (_Float16)(f - (float)h);
      f = xb4[e] - mub[e8 + 4 + e];
      h = (_Float16)f;
      sXhi[e8 + 4 + e][kk] = h; sXlo[e8 + 4 + e][kk] = (_Float16)(f - (float)h);
    }
    __syncthreads();

    v16h ahi = retf_frag_a(sQhi, ti * 16 + m, kh);
    v16h alo = retf_frag_a(sQlo, ti * 16 + m, kh);
    v16h b0h = retf_frag_b(sXhi, tj0 * 16 + m, kh);
    v16h b0l = retf_frag_b(sXlo, tj0 * 16 + m, kh);
    v16h b1h = retf_frag_b(sXhi, tj1 * 16 + m, kh);
    v16h b1l = retf_frag_b(sXlo, tj1 * 16 + m, kh);

    acc0 = WMMA_F16(ahi, b0h, acc0);
    acc0 = WMMA_F16(ahi, b0l, acc0);
    acc0 = WMMA_F16(alo, b0h, acc0);
    acc1 = WMMA_F16(ahi, b1h, acc1);
    acc1 = WMMA_F16(ahi, b1l, acc1);
    acc1 = WMMA_F16(alo, b1h, acc1);
  }

  float s0 = 0.f, s1 = 0.f;
#pragma unroll
  for (int r8 = 0; r8 < 8; ++r8) {
    s0 += acc0[r8] * acc0[r8];
    s1 += acc1[r8] * acc1[r8];
  }
  sAcc[tj0 * 16 + m][ti * 2 + kh] = s0;
  sAcc[tj1 * 16 + m][ti * 2 + kh] = s1;
  __syncthreads();
  if (tid < 64) {
    float v = 0.f;
#pragma unroll
    for (int j = 0; j < 8; ++j) v += sAcc[tid][j];
    Iout[(size_t)(mod * Bn + b) * Dn + d0 + tid] = v;
  }
}

// ---------------------- K6: radix-select bottom-410 ------------------------
__global__ __launch_bounds__(256)
void retf_select_kernel(const float* __restrict__ I, float* __restrict__ Mout) {
  int b = blockIdx.x, mod = blockIdx.y;
  const float* Ib = I + (size_t)(mod * Bn + b) * Dn;
  float* Mb = Mout + (size_t)(mod * Bn + b) * Dn;

  __shared__ unsigned sKey[Dn];
  __shared__ unsigned sHist[256];
  __shared__ unsigned sState[3];   // prefix, k remaining, tie counter

  int t = threadIdx.x;
  for (int i = t; i < Dn; i += 256) {
    unsigned u = __float_as_uint(Ib[i]);
    sKey[i] = (u >> 31) ? ~u : (u | 0x80000000u);   // order-preserving key
  }
  if (t == 0) { sState[0] = 0u; sState[1] = KLOW; sState[2] = 0u; }
  __syncthreads();

  for (int rnd = 0; rnd < 4; ++rnd) {
    int shift = 24 - 8 * rnd;
    unsigned prefix = sState[0];
    unsigned hmask = (rnd == 0) ? 0u : (0xFFFFFFFFu << (shift + 8));
    sHist[t] = 0u;
    __syncthreads();
    for (int i = t; i < Dn; i += 256) {
      unsigned k = sKey[i];
      if ((k & hmask) == prefix) atomicAdd(&sHist[(k >> shift) & 255u], 1u);
    }
    __syncthreads();
    if (t == 0) {
      unsigned kk = sState[1], cum = 0u;
      int bsel = 255;
      for (int bin = 0; bin < 256; ++bin) {
        unsigned c = sHist[bin];
        if (cum + c >= kk) { bsel = bin; break; }
        cum += c;
      }
      sState[0] = prefix | ((unsigned)bsel << shift);
      sState[1] = kk - cum;
    }
    __syncthreads();
  }
  unsigned T = sState[0];
  unsigned kk = sState[1];
  for (int i = t; i < Dn; i += 256) {
    unsigned k = sKey[i];
    float mv = 0.f;
    if (k < T) mv = 1.f;
    else if (k == T) { if (atomicAdd(&sState[2], 1u) < kk) mv = 1.f; }
    Mb[i] = mv;
  }
}

// ------------------------------ K7: fuse -----------------------------------
__global__ void retf_fuse_kernel(const float* __restrict__ X, const float* __restrict__ pool,
                                 const float* __restrict__ alpha, const float* __restrict__ msk,
                                 float* __restrict__ out, int L, long total4) {
  long i = (long)blockIdx.x * blockDim.x + threadIdx.x;
  if (i >= total4) return;
  size_t e = (size_t)i * 4;
  int d = (int)(e & (Dn - 1));
  int b = (int)(e / ((size_t)L * Dn));
  v4f x  = __builtin_nontemporal_load((const v4f*)(X + e));
  v4f p  = *(const v4f*)(pool + (size_t)b * Dn + d);
  v4f a  = *(const v4f*)(alpha + d);
  v4f mm = *(const v4f*)(msk + (size_t)b * Dn + d);
  v4f o;
  o.x = x.x + mm.x * a.x * (p.x - x.x);
  o.y = x.y + mm.y * a.y * (p.y - x.y);
  o.z = x.z + mm.z * a.z * (p.z - x.z);
  o.w = x.w + mm.w * a.w * (p.w - x.w);
  __builtin_nontemporal_store(o, (v4f*)(out + e));
}

// ------------------------------ launcher -----------------------------------
extern "C" void kernel_launch(void* const* d_in, const int* in_sizes, int n_in,
                              void* d_out, int out_size, void* d_ws, size_t ws_size,
                              hipStream_t stream) {
  (void)in_sizes; (void)n_in; (void)out_size; (void)ws_size;
  const float* tx = (const float*)d_in[0];
  const float* ix = (const float*)d_in[1];
  const unsigned char* tmask = (const unsigned char*)d_in[2];
  const float* aimg = (const float*)d_in[3];
  const float* atxt = (const float*)d_in[4];

  // workspace layout (floats)
  const size_t OFF_POOL = 0;                                   // [2][B][D]
  const size_t OFF_MU   = OFF_POOL + (size_t)2 * Bn * Dn;      // [2][B][D]
  const size_t OFF_R    = OFF_MU   + (size_t)2 * Bn * Dn;      // [2][B][257]
  const size_t OFF_G    = OFF_R    + (size_t)2 * Bn * 257;     // [2][B][256*256]
  const size_t OFF_Q    = OFF_G    + (size_t)2 * Bn * Sn * Sn; // [2][B][256*64]
  const size_t OFF_I    = OFF_Q    + (size_t)2 * Bn * Sn * Qr; // [2][B][D]
  const size_t OFF_M    = OFF_I    + (size_t)2 * Bn * Dn;      // [2][B][D]
  const size_t OFF_A    = OFF_M    + (size_t)2 * Bn * Dn;      // [2][D]

  float* ws   = (float*)d_ws;
  float* pool = ws + OFF_POOL;
  float* mu   = ws + OFF_MU;
  float* rr   = ws + OFF_R;
  float* G    = ws + OFF_G;
  float* Qg   = ws + OFF_Q;
  float* I    = ws + OFF_I;
  float* M    = ws + OFF_M;
  float* A    = ws + OFF_A;

  retf_alpha_kernel<<<dim3(Dn / 256, 2), 256, 0, stream>>>(atxt, aimg, A);
  retf_pool_mu_kernel<<<dim3(Dn / 256, Bn), 256, 0, stream>>>(tx, tmask, LT, pool, mu);
  retf_pool_mu_kernel<<<dim3(Dn / 256, Bn), 256, 0, stream>>>(ix, nullptr, LI,
                                                              pool + (size_t)Bn * Dn,
                                                              mu + (size_t)Bn * Dn);
  retf_rdot_kernel<<<dim3(257, Bn, 2), 256, 0, stream>>>(tx, ix, mu, rr);
  retf_gram_kernel<<<dim3(16, Bn, 2), 256, 0, stream>>>(tx, ix, rr, G);
  retf_subspace_kernel<<<dim3(Bn, 2), 256, 0, stream>>>(G, Qg);
  retf_importance_kernel<<<dim3(Dn / 64, Bn, 2), 256, 0, stream>>>(tx, ix, mu, Qg, I);
  retf_select_kernel<<<dim3(Bn, 2), 256, 0, stream>>>(I, M);

  long t4 = (long)Bn * LT * Dn / 4;
  retf_fuse_kernel<<<dim3((unsigned)((t4 + 255) / 256)), 256, 0, stream>>>(
      tx, pool + (size_t)Bn * Dn, A, M, (float*)d_out, LT, t4);
  long i4 = (long)Bn * LI * Dn / 4;
  retf_fuse_kernel<<<dim3((unsigned)((i4 + 255) / 256)), 256, 0, stream>>>(
      ix, pool, A + Dn, M + (size_t)Bn * Dn,
      (float*)d_out + (size_t)Bn * LT * Dn, LI, i4);
}
